// MambaBlock_5308579578199
// MI455X (gfx1250) — matrix operand order repaired and checked by reading, not compiled, not run on hardware
//
#include <hip/hip_runtime.h>

// ---------------------------------------------------------------------------
// Mamba block on gfx1250 (MI455X):
//  - BF16 WMMA (v_wmma_f32_16x16x32_bf16) for in_proj / out_proj GEMMs,
//    2Mx4N register blocking per wave, global_prefetch on the A stream.
//  - Chunked (Blelloch-style) selective scan: 32 chunks x 2048 channels of
//    affine summaries -> tiny serial combine -> parallel replay.
// ---------------------------------------------------------------------------

#define D_MODEL 1024
#define D_STATE 16
#define D_CONV  4
#define D_INNER 2048
#define SEQ     2048
#define NCHUNK  32
#define CLEN    (SEQ / NCHUNK)   // 64

typedef unsigned short u16t;
typedef __attribute__((ext_vector_type(16))) __bf16 v16bf;
typedef __attribute__((ext_vector_type(8)))  float  v8f;

union ABfrag { v16bf v; uint4 u[2]; };

__device__ __forceinline__ u16t f2bf(float f) {
    unsigned u = __float_as_uint(f);
    unsigned r = u + 0x7FFFu + ((u >> 16) & 1u);   // round-to-nearest-even
    return (u16t)(r >> 16);
}

__device__ __forceinline__ float softplus_f(float x) {
    return (x > 20.f) ? x : __logf(1.f + __expf(x));
}

// --- fp32 -> bf16 conversion ------------------------------------------------
__global__ __launch_bounds__(256)
void cvt_bf16_k(const float* __restrict__ src, u16t* __restrict__ dst, int n) {
    int i = blockIdx.x * 256 + threadIdx.x;
    if (i < n) dst[i] = f2bf(src[i]);
}

// --- LayerNorm over D_MODEL, emit bf16 activations --------------------------
__global__ __launch_bounds__(256)
void ln_bf16_k(const float* __restrict__ x, const float* __restrict__ g,
               const float* __restrict__ b, u16t* __restrict__ xn) {
    int s = blockIdx.x, t = threadIdx.x;
    __shared__ float red[256];
    const float* xr = x + (size_t)s * D_MODEL;

    float sum = 0.f;
    for (int d = t; d < D_MODEL; d += 256) sum += xr[d];
    red[t] = sum; __syncthreads();
    for (int o = 128; o > 0; o >>= 1) { if (t < o) red[t] += red[t + o]; __syncthreads(); }
    float mu = red[0] * (1.f / D_MODEL);
    __syncthreads();

    float vs = 0.f;
    for (int d = t; d < D_MODEL; d += 256) { float df = xr[d] - mu; vs += df * df; }
    red[t] = vs; __syncthreads();
    for (int o = 128; o > 0; o >>= 1) { if (t < o) red[t] += red[t + o]; __syncthreads(); }
    float rstd = rsqrtf(red[0] * (1.f / D_MODEL) + 1e-5f);

    for (int d = t; d < D_MODEL; d += 256) {
        float v = (xr[d] - mu) * rstd * g[d] + b[d];
        xn[(size_t)s * D_MODEL + d] = f2bf(v);
    }
}

// --- BF16 WMMA GEMM: Out[M,N] = A[M,K] * B[N,K]^T (+ Cres), fp32 accum ------
// One wave per block; wave owns a 32x64 strip (2 M-tiles x 4 N-tiles):
// each B fragment feeds two WMMAs, halving weight traffic from L2.
__global__ __launch_bounds__(32)
void wmma_gemm_bf16_k(const u16t* __restrict__ A, const u16t* __restrict__ B,
                      const float* __restrict__ Cres, float* __restrict__ Out,
                      int K, int lda, int ldb, int ldc) {
    const int lane   = threadIdx.x;       // 0..31 (wave32)
    const int half   = lane >> 4;
    const int mq     = lane & 15;
    const int nt0    = blockIdx.x * 4;    // four 16-col tiles
    const int mtile0 = blockIdx.y * 2;    // two 16-row tiles

    v8f acc[2][4];
    if (Cres) {
        for (int m = 0; m < 2; ++m)
            for (int t = 0; t < 4; ++t)
                for (int r = 0; r < 8; ++r)
                    acc[m][t][r] = Cres[(size_t)((mtile0 + m) * 16 + r + 8 * half) * ldc +
                                        (nt0 + t) * 16 + mq];
    } else {
        for (int m = 0; m < 2; ++m)
            for (int t = 0; t < 4; ++t)
                for (int r = 0; r < 8; ++r) acc[m][t][r] = 0.f;
    }

    const u16t* arow0 = A + (size_t)(mtile0 * 16 + mq) * lda;
    const u16t* arow1 = arow0 + (size_t)16 * lda;
    for (int kb = 0; kb < K; kb += 32) {
        // A 16x32 frag: lane<16 -> K{0..7,16..23}; lane>=16 -> K{8..15,24..31}
        ABfrag a0, a1;
        const u16t* ap0 = arow0 + kb + half * 8;
        const u16t* ap1 = arow1 + kb + half * 8;
        a0.u[0] = *(const uint4*)(ap0);
        a0.u[1] = *(const uint4*)(ap0 + 16);
        a1.u[0] = *(const uint4*)(ap1);
        a1.u[1] = *(const uint4*)(ap1 + 16);
        __builtin_prefetch((const void*)(ap0 + 64), 0, 0);   // global_prefetch_b8
        __builtin_prefetch((const void*)(ap1 + 64), 0, 0);

        for (int t = 0; t < 4; ++t) {
            // B 32x16 frag: lane holds col (lane&15); lane<16 -> K{0..7,16..23}; lane>=16 -> K{8..15,24..31}
            const u16t* bp = B + (size_t)((nt0 + t) * 16 + mq) * ldb + kb + half * 8;
            ABfrag bf;
            bf.u[0] = *(const uint4*)(bp);
            bf.u[1] = *(const uint4*)(bp + 16);
            acc[0][t] = __builtin_amdgcn_wmma_f32_16x16x32_bf16(
                            false, a0.v, false, bf.v, (short)0, acc[0][t], false, false);
            acc[1][t] = __builtin_amdgcn_wmma_f32_16x16x32_bf16(
                            false, a1.v, false, bf.v, (short)0, acc[1][t], false, false);
        }
    }

    for (int m = 0; m < 2; ++m)
        for (int t = 0; t < 4; ++t)
            for (int r = 0; r < 8; ++r)
                Out[(size_t)((mtile0 + m) * 16 + r + 8 * half) * ldc +
                    (nt0 + t) * 16 + mq] = acc[m][t][r];
}

// --- depthwise causal conv (width 4) + SiLU ---------------------------------
__global__ __launch_bounds__(256)
void conv_silu_k(const float* __restrict__ xz, const float* __restrict__ cw,
                 const float* __restrict__ cb, float* __restrict__ xs) {
    int i = blockIdx.x * 256 + threadIdx.x;     // over SEQ*D_INNER
    int s = i >> 11, c = i & (D_INNER - 1);
    float acc = cb[c];
    #pragma unroll
    for (int j = 0; j < D_CONV; ++j) {
        int sp = s - (D_CONV - 1) + j;
        if (sp >= 0) acc += xz[(size_t)sp * (2 * D_INNER) + c] * cw[c * D_CONV + j];
    }
    xs[i] = acc / (1.f + __expf(-acc));         // silu
}

// --- x_proj: xp[s, 0..32] = xs[s,:] . xproj_w[e,:]  (tiny GEMM, VALU) -------
__global__ __launch_bounds__(256)
void xproj_k(const float* __restrict__ xs, const float* __restrict__ w,
             float* __restrict__ xp) {
    int s = blockIdx.x, t = threadIdx.x;
    const int NE = 2 * D_STATE + 1;             // 33
    float acc[NE];
    #pragma unroll
    for (int e = 0; e < NE; ++e) acc[e] = 0.f;
    const float* xr = xs + (size_t)s * D_INNER;
    for (int c = t; c < D_INNER; c += 256) {
        float xv = xr[c];
        #pragma unroll
        for (int e = 0; e < NE; ++e) acc[e] += xv * w[e * D_INNER + c];
    }
    __shared__ float red[NE][256];
    #pragma unroll
    for (int e = 0; e < NE; ++e) red[e][t] = acc[e];
    __syncthreads();
    for (int o = 128; o > 0; o >>= 1) {
        if (t < o)
            #pragma unroll
            for (int e = 0; e < NE; ++e) red[e][t] += red[e][t + o];
        __syncthreads();
    }
    if (t < NE) xp[(size_t)s * NE + t] = red[t][0];
}

// --- scan phase A: per-(channel, chunk) affine summary h_end = P*h0 + S -----
__global__ __launch_bounds__(256)
void scan_chunk_k(const float* __restrict__ xp, const float* __restrict__ xs,
                  const float* __restrict__ dt_w, const float* __restrict__ dt_b,
                  const float* __restrict__ A_log,
                  float* __restrict__ sumP, float* __restrict__ sumS) {
    int i = blockIdx.x * 256 + threadIdx.x;     // over D_INNER*NCHUNK
    int c = i & (D_INNER - 1);
    int chunk = i >> 11;
    float P[D_STATE], S[D_STATE], An[D_STATE];
    #pragma unroll
    for (int n = 0; n < D_STATE; ++n) { P[n] = 1.f; S[n] = 0.f; An[n] = -__expf(A_log[n]); }
    const float dtw = dt_w[c], dtb = dt_b[c];

    for (int s = chunk * CLEN; s < (chunk + 1) * CLEN; ++s) {
        const float* xpr = xp + (size_t)s * 33;
        float del = softplus_f(xpr[0] * dtw + dtb);
        float dx  = del * xs[(size_t)s * D_INNER + c];
        #pragma unroll
        for (int n = 0; n < D_STATE; ++n) {
            float dA = __expf(del * An[n]);
            P[n] *= dA;
            S[n] = dA * S[n] + dx * xpr[1 + n];
        }
    }
    float* pd = sumP + ((size_t)chunk * D_INNER + c) * D_STATE;
    float* sd = sumS + ((size_t)chunk * D_INNER + c) * D_STATE;
    #pragma unroll
    for (int n = 0; n < D_STATE; ++n) { pd[n] = P[n]; sd[n] = S[n]; }
}

// --- scan phase B: serial combine over 32 chunk summaries per channel -------
__global__ __launch_bounds__(256)
void scan_combine_k(const float* __restrict__ sumP, const float* __restrict__ sumS,
                    float* __restrict__ hinit) {
    int c = blockIdx.x * 256 + threadIdx.x;     // channel
    float h[D_STATE];
    #pragma unroll
    for (int n = 0; n < D_STATE; ++n) h[n] = 0.f;
    for (int chunk = 0; chunk < NCHUNK; ++chunk) {
        size_t base = ((size_t)chunk * D_INNER + c) * D_STATE;
        float* hd = hinit + base;
        #pragma unroll
        for (int n = 0; n < D_STATE; ++n) {
            hd[n] = h[n];
            h[n] = sumP[base + n] * h[n] + sumS[base + n];
        }
    }
}

// --- scan phase C: parallel replay with known chunk-initial state -----------
__global__ __launch_bounds__(256)
void scan_out_k(const float* __restrict__ xp, const float* __restrict__ xs,
                const float* __restrict__ xz, const float* __restrict__ dt_w,
                const float* __restrict__ dt_b, const float* __restrict__ A_log,
                const float* __restrict__ Dv, const float* __restrict__ hinit,
                u16t* __restrict__ ybf) {
    int i = blockIdx.x * 256 + threadIdx.x;     // over D_INNER*NCHUNK
    int c = i & (D_INNER - 1);
    int chunk = i >> 11;
    float h[D_STATE], An[D_STATE];
    const float* hi = hinit + ((size_t)chunk * D_INNER + c) * D_STATE;
    #pragma unroll
    for (int n = 0; n < D_STATE; ++n) { h[n] = hi[n]; An[n] = -__expf(A_log[n]); }
    const float dtw = dt_w[c], dtb = dt_b[c], Dc = Dv[c];

    for (int s = chunk * CLEN; s < (chunk + 1) * CLEN; ++s) {
        const float* xpr = xp + (size_t)s * 33;
        float del = softplus_f(xpr[0] * dtw + dtb);
        float xv  = xs[(size_t)s * D_INNER + c];
        float dx  = del * xv;
        float y   = 0.f;
        #pragma unroll
        for (int n = 0; n < D_STATE; ++n) {
            float dA = __expf(del * An[n]);
            h[n] = dA * h[n] + dx * xpr[1 + n];      // Bm
            y   += h[n] * xpr[17 + n];               // Cm
        }
        y += xv * Dc;
        float zv = xz[(size_t)s * (2 * D_INNER) + D_INNER + c];
        y *= zv / (1.f + __expf(-zv));               // gate with silu(z)
        ybf[(size_t)s * D_INNER + c] = f2bf(y);
    }
}

// ---------------------------------------------------------------------------
extern "C" void kernel_launch(void* const* d_in, const int* in_sizes, int n_in,
                              void* d_out, int out_size, void* d_ws, size_t ws_size,
                              hipStream_t stream) {
    (void)in_sizes; (void)n_in; (void)out_size; (void)ws_size;
    const float* x       = (const float*)d_in[0];
    const float* ln_g    = (const float*)d_in[1];
    const float* ln_b    = (const float*)d_in[2];
    const float* in_w    = (const float*)d_in[3];
    const float* conv_w  = (const float*)d_in[4];
    const float* conv_b  = (const float*)d_in[5];
    const float* xproj_w = (const float*)d_in[6];
    const float* dt_w    = (const float*)d_in[7];
    const float* dt_b    = (const float*)d_in[8];
    const float* A_log   = (const float*)d_in[9];
    const float* Dvec    = (const float*)d_in[10];
    const float* out_w   = (const float*)d_in[11];
    float* out = (float*)d_out;

    char* ws = (char*)d_ws;
    size_t o_xn   = 0;                                            // SEQ*D_MODEL bf16
    size_t o_inw  = o_xn  + (size_t)SEQ * D_MODEL * 2;
    size_t o_outw = o_inw + (size_t)2 * D_INNER * D_MODEL * 2;
    size_t o_xz   = o_outw + (size_t)D_MODEL * D_INNER * 2;
    size_t o_xs   = o_xz  + (size_t)SEQ * 2 * D_INNER * 4;
    size_t o_xp   = o_xs  + (size_t)SEQ * D_INNER * 4;
    size_t o_y    = o_xp  + (size_t)SEQ * 33 * 4;
    size_t o_sp   = o_y   + (size_t)SEQ * D_INNER * 2;            // chunk P
    size_t o_ss   = o_sp  + (size_t)NCHUNK * D_INNER * D_STATE * 4;
    size_t o_hi   = o_ss  + (size_t)NCHUNK * D_INNER * D_STATE * 4;

    u16t*  xn_bf   = (u16t*)(ws + o_xn);
    u16t*  inw_bf  = (u16t*)(ws + o_inw);
    u16t*  outw_bf = (u16t*)(ws + o_outw);
    float* xz      = (float*)(ws + o_xz);
    float* xs      = (float*)(ws + o_xs);
    float* xp      = (float*)(ws + o_xp);
    u16t*  y_bf    = (u16t*)(ws + o_y);
    float* sumP    = (float*)(ws + o_sp);
    float* sumS    = (float*)(ws + o_ss);
    float* hinit   = (float*)(ws + o_hi);

    // 1) weights -> bf16
    {
        int n1 = 2 * D_INNER * D_MODEL;
        cvt_bf16_k<<<(n1 + 255) / 256, 256, 0, stream>>>(in_w, inw_bf, n1);
        int n2 = D_MODEL * D_INNER;
        cvt_bf16_k<<<(n2 + 255) / 256, 256, 0, stream>>>(out_w, outw_bf, n2);
    }
    // 2) layernorm -> bf16 activations
    ln_bf16_k<<<SEQ, 256, 0, stream>>>(x, ln_g, ln_b, xn_bf);
    // 3) in_proj GEMM: xz[SEQ, 2*D_INNER]  (M=2048, N=4096, K=1024)
    wmma_gemm_bf16_k<<<dim3((2 * D_INNER) / 64, SEQ / 32), 32, 0, stream>>>(
        xn_bf, inw_bf, nullptr, xz, D_MODEL, D_MODEL, D_MODEL, 2 * D_INNER);
    // 4) depthwise conv + silu -> xs
    conv_silu_k<<<(SEQ * D_INNER) / 256, 256, 0, stream>>>(xz, conv_w, conv_b, xs);
    // 5) x_proj (tiny)
    xproj_k<<<SEQ, 256, 0, stream>>>(xs, xproj_w, xp);
    // 6) chunked selective scan
    scan_chunk_k<<<(D_INNER * NCHUNK) / 256, 256, 0, stream>>>(
        xp, xs, dt_w, dt_b, A_log, sumP, sumS);
    scan_combine_k<<<D_INNER / 256, 256, 0, stream>>>(sumP, sumS, hinit);
    scan_out_k<<<(D_INNER * NCHUNK) / 256, 256, 0, stream>>>(
        xp, xs, xz, dt_w, dt_b, A_log, Dvec, hinit, y_bf);
    // 7) out_proj GEMM + residual: out = y @ out_w^T + x  (M=2048, N=1024, K=2048)
    wmma_gemm_bf16_k<<<dim3(D_MODEL / 64, SEQ / 32), 32, 0, stream>>>(
        y_bf, outw_bf, x, out, D_INNER, D_INNER, D_INNER, D_MODEL);
}
